// SlidingWindowPyramidAttention_71227737637162
// MI455X (gfx1250) — compile-verified, matrix operand order checked by reading, not run on hardware
//
#include <hip/hip_runtime.h>
#include <cstdint>
#include <cstddef>

// ---------------------------------------------------------------------------
// Types for CDNA5 WMMA (wave32)
// ---------------------------------------------------------------------------
typedef __bf16 bf16_t;
typedef __attribute__((ext_vector_type(16))) __bf16 v16bf;
typedef __attribute__((ext_vector_type(8)))  float  v8f;

__device__ __forceinline__ v8f wmma_bf(v16bf a, v16bf b, v8f c) {
    // D = A(16x32 bf16) * B(32x16 bf16) + C(16x16 f32)
    return __builtin_amdgcn_wmma_f32_16x16x32_bf16(
        /*neg_a=*/false, a, /*neg_b=*/false, b,
        /*c_mod=*/(short)0, c, /*reuse_a=*/false, /*reuse_b=*/false);
}

// K index of the low element of VGPR-pair j (0..7) for lane-half 'half' (0/1).
// ISA 7.12.2: VGPR0..3 hold K={0..7}+half*8, VGPR4..7 hold K={16..23}+half*8.
__device__ __forceinline__ int kidx(int j, int half) {
    return ((j & 3) << 1) + (half << 3) + ((j >> 2) << 4);
}

// window position -> flat position p = b*4096 + y*64 + x
__device__ __forceinline__ int wpos(int b, int wy, int wx, int s) {
    int y = (wy << 3) + (s >> 3);
    int x = (wx << 3) + (s & 7);
    return (b << 12) + (y << 6) + x;
}

// ---------------------------------------------------------------------------
// Pack weight W[N][K] (row-major fp32, out = X @ W^T) into WMMA B-fragment
// order: [(kTile*nTiles+nTile)][lane][16] bf16, 32B contiguous per lane.
// ---------------------------------------------------------------------------
__global__ void pack_weight(const float* __restrict__ W, bf16_t* __restrict__ P,
                            int K, int N) {
    int frag = blockIdx.x;           // kTile * nTiles + nTile
    int lane = threadIdx.x;          // 0..31
    int nTiles = N >> 4;
    int kTile = frag / nTiles;
    int nTile = frag - kTile * nTiles;
    int o = (nTile << 4) + (lane & 15);   // output channel (B column)
    int half = lane >> 4;
    bf16_t* dst = P + (((size_t)frag * 32 + lane) << 4);
#pragma unroll
    for (int j = 0; j < 8; ++j) {
        int kb = (kTile << 5) + kidx(j, half);
        dst[2 * j]     = (bf16_t)W[(size_t)o * K + kb];
        dst[2 * j + 1] = (bf16_t)W[(size_t)o * K + kb + 1];
    }
}

// ---------------------------------------------------------------------------
// GEMM, A row-major bf16 [M][K], B packed fragments, bias per column.
// Each wave computes a 16 x (16*NB) strip: A fragment loaded once per k-step,
// NB independent WMMAs issued back-to-back (no D->C RAW between them).
// ---------------------------------------------------------------------------
template <int NB, bool OUT_BF16>
__global__ void gemm_rowA(const bf16_t* __restrict__ A, const bf16_t* __restrict__ Bp,
                          const float* __restrict__ bias,
                          float* __restrict__ D, bf16_t* __restrict__ D16,
                          int M, int K, int N) {
    int wave = threadIdx.x >> 5;
    int lane = threadIdx.x & 31;
    int nTiles = N >> 4;
    int nGroups = nTiles / NB;
    int mTiles = M >> 4;
    int strip = blockIdx.x * (blockDim.x >> 5) + wave;
    if (strip >= mTiles * nGroups) return;
    int mTile = strip / nGroups, nGroup = strip - mTile * nGroups;
    int mBase = mTile << 4;
    int nTile0 = nGroup * NB;
    int row = mBase + (lane & 15);
    int half = lane >> 4;

    v8f acc[NB];
#pragma unroll
    for (int nb = 0; nb < NB; ++nb) acc[nb] = (v8f){0.f,0.f,0.f,0.f,0.f,0.f,0.f,0.f};

    int kTiles = K >> 5;
    for (int kk = 0; kk < kTiles; ++kk) {
        union { v16bf v; uint32_t u[8]; } af;
        const bf16_t* Ar = A + (size_t)row * K + (kk << 5);
#pragma unroll
        for (int j = 0; j < 8; ++j)
            af.u[j] = *(const uint32_t*)(Ar + kidx(j, half));
        const bf16_t* Bbase = Bp + ((((size_t)kk * nTiles + nTile0) * 32 + lane) << 4);
#pragma unroll
        for (int nb = 0; nb < NB; ++nb) {
            v16bf bfrag = *(const v16bf*)(Bbase + ((size_t)nb << 9));  // 32 lanes*16 el
            acc[nb] = wmma_bf(af.v, bfrag, acc[nb]);
        }
    }
#pragma unroll
    for (int nb = 0; nb < NB; ++nb) {
        int col = ((nTile0 + nb) << 4) + (lane & 15);
        float bb = bias ? bias[col] : 0.f;
#pragma unroll
        for (int r = 0; r < 8; ++r) {
            int m = mBase + r + (half << 3);
            float v = acc[nb][r] + bb;
            if (OUT_BF16) D16[(size_t)m * N + col] = (bf16_t)v;
            else          D[(size_t)m * N + col]   = v;
        }
    }
}

// ---------------------------------------------------------------------------
// GEMM with A read from channel-major fp32 [256][4096] (one B- or B,L-slice of
// a [.,C,H,W] tensor), optional per-channel additive term (level_embed).
// M = 4096 rows (positions in slice), K = 256. Output row = rowOffset + m.
// N-blocked like gemm_rowA (A fragment built once per k-step).
// ---------------------------------------------------------------------------
template <int NB, bool OUT_BF16>
__global__ void gemm_chw(const float* __restrict__ Asrc, const float* __restrict__ chanAdd,
                         const bf16_t* __restrict__ Bp, const float* __restrict__ bias,
                         float* __restrict__ D, bf16_t* __restrict__ D16,
                         int rowOffset, int N) {
    const int K = 256;
    int wave = threadIdx.x >> 5;
    int lane = threadIdx.x & 31;
    int nTiles = N >> 4;
    int nGroups = nTiles / NB;
    int strip = blockIdx.x * (blockDim.x >> 5) + wave;
    if (strip >= 256 * nGroups) return;          // 4096/16 = 256 mTiles
    int mTile = strip / nGroups, nGroup = strip - mTile * nGroups;
    int mBase = mTile << 4;
    int nTile0 = nGroup * NB;
    int m = mBase + (lane & 15);
    int half = lane >> 4;

    v8f acc[NB];
#pragma unroll
    for (int nb = 0; nb < NB; ++nb) acc[nb] = (v8f){0.f,0.f,0.f,0.f,0.f,0.f,0.f,0.f};

    for (int kk = 0; kk < K / 32; ++kk) {
        union { v16bf v; __bf16 h[16]; } af;
#pragma unroll
        for (int j = 0; j < 8; ++j) {
            int c = (kk << 5) + kidx(j, half);
            float a0 = Asrc[(size_t)c * 4096 + m];
            float a1 = Asrc[(size_t)(c + 1) * 4096 + m];
            if (chanAdd) { a0 += chanAdd[c]; a1 += chanAdd[c + 1]; }
            af.h[2 * j]     = (bf16_t)a0;
            af.h[2 * j + 1] = (bf16_t)a1;
        }
        const bf16_t* Bbase = Bp + ((((size_t)kk * nTiles + nTile0) * 32 + lane) << 4);
#pragma unroll
        for (int nb = 0; nb < NB; ++nb) {
            v16bf bfrag = *(const v16bf*)(Bbase + ((size_t)nb << 9));
            acc[nb] = wmma_bf(af.v, bfrag, acc[nb]);
        }
    }
#pragma unroll
    for (int nb = 0; nb < NB; ++nb) {
        int col = ((nTile0 + nb) << 4) + (lane & 15);
        float bb = bias ? bias[col] : 0.f;
#pragma unroll
        for (int r = 0; r < 8; ++r) {
            int mm = mBase + r + (half << 3);
            size_t rp = (size_t)(rowOffset + mm);
            float v = acc[nb][r] + bb;
            if (OUT_BF16) D16[rp * N + col] = (bf16_t)v;
            else          D[rp * N + col]   = v;
        }
    }
}

// ---------------------------------------------------------------------------
// Final GEMM: A = attn rows bf16 [16384][256], B = packed op_w, store fp32
// into [B=4][C=256][4096] with bias. M=16384, K=256, N=256, NB=4.
// ---------------------------------------------------------------------------
__global__ void gemm_op(const bf16_t* __restrict__ A, const bf16_t* __restrict__ Bp,
                        const float* __restrict__ bias, float* __restrict__ Out) {
    const int K = 256, N = 256, NB = 4;
    int wave = threadIdx.x >> 5;
    int lane = threadIdx.x & 31;
    int nTiles = N >> 4;            // 16
    int nGroups = nTiles / NB;      // 4
    int strip = blockIdx.x * (blockDim.x >> 5) + wave;
    if (strip >= 1024 * nGroups) return;
    int mTile = strip / nGroups, nGroup = strip - mTile * nGroups;
    int mBase = mTile << 4;
    int nTile0 = nGroup * NB;
    int row = mBase + (lane & 15);
    int half = lane >> 4;

    v8f acc[NB];
#pragma unroll
    for (int nb = 0; nb < NB; ++nb) acc[nb] = (v8f){0.f,0.f,0.f,0.f,0.f,0.f,0.f,0.f};

    for (int kk = 0; kk < K / 32; ++kk) {
        union { v16bf v; uint32_t u[8]; } af;
        const bf16_t* Ar = A + (size_t)row * K + (kk << 5);
#pragma unroll
        for (int j = 0; j < 8; ++j)
            af.u[j] = *(const uint32_t*)(Ar + kidx(j, half));
        const bf16_t* Bbase = Bp + ((((size_t)kk * nTiles + nTile0) * 32 + lane) << 4);
#pragma unroll
        for (int nb = 0; nb < NB; ++nb) {
            v16bf bfrag = *(const v16bf*)(Bbase + ((size_t)nb << 9));
            acc[nb] = wmma_bf(af.v, bfrag, acc[nb]);
        }
    }
#pragma unroll
    for (int nb = 0; nb < NB; ++nb) {
        int col = ((nTile0 + nb) << 4) + (lane & 15);
        float bb = bias[col];
#pragma unroll
        for (int r = 0; r < 8; ++r) {
            int p = mBase + r + (half << 3);   // flat position
            int b = p >> 12;
            int hw = p & 4095;
            Out[(((size_t)b * 256 + col) << 12) + hw] = acc[nb][r] + bb;
        }
    }
}

// ---------------------------------------------------------------------------
// Per-window GroupNorm(8 groups over 128 ch) + ReLU for both heads; bf16 out.
// Block = one window (256 blocks), 128 threads (thread == channel).
// ---------------------------------------------------------------------------
__global__ void gn_relu(const float* __restrict__ Hso, const float* __restrict__ Haw,
                        const float* __restrict__ g_so, const float* __restrict__ b_so,
                        const float* __restrict__ g_aw, const float* __restrict__ b_aw,
                        bf16_t* __restrict__ Oso, bf16_t* __restrict__ Oaw) {
    int win = blockIdx.x;
    int b = win >> 6, wi = win & 63, wy = wi >> 3, wx = wi & 7;
    int t = threadIdx.x;            // 0..127 == channel
    int gidx = t >> 4, laneg = t & 15;
    __shared__ float psum[128], psq[128];
    __shared__ float s_mu[8], s_ri[8];

    for (int pass = 0; pass < 2; ++pass) {
        const float* H = pass ? Haw : Hso;
        const float* gamma = pass ? g_aw : g_so;
        const float* beta  = pass ? b_aw : b_so;
        bf16_t* O = pass ? Oaw : Oso;

        float sum = 0.f, sq = 0.f;
        for (int s = 0; s < 64; ++s) {
            int p = wpos(b, wy, wx, s);
            float v = H[((size_t)p << 7) + t];
            sum += v; sq += v * v;
        }
        psum[t] = sum; psq[t] = sq;
        __syncthreads();
        if (laneg == 0) {
            float S = 0.f, Q = 0.f;
            for (int i = 0; i < 16; ++i) { S += psum[(gidx << 4) + i]; Q += psq[(gidx << 4) + i]; }
            float mu = S * (1.f / 1024.f);
            float var = Q * (1.f / 1024.f) - mu * mu;
            s_mu[gidx] = mu;
            s_ri[gidx] = rsqrtf(var + 1e-5f);
        }
        __syncthreads();
        float mu = s_mu[gidx], ri = s_ri[gidx];
        float gg = gamma[t], bb = beta[t];
        for (int s = 0; s < 64; ++s) {
            int p = wpos(b, wy, wx, s);
            float v = (H[((size_t)p << 7) + t] - mu) * ri * gg + bb;
            v = v > 0.f ? v : 0.f;
            O[((size_t)p << 7) + t] = (bf16_t)v;
        }
        __syncthreads();
    }
}

// ---------------------------------------------------------------------------
// Deformable sampling + softmax + attention combine.
// Block = (window, head): 2048 blocks, 64 threads (thread == query s).
// off  : [p][192] fp32, channel ((h*3+l)*4+pp)*2 + {x,y}
// awl  : [p][96]  fp32 logits, channel h*12 + l*4 + pp
// Vproj: [l][p][256] bf16 (channel h*32+d)
// out  : attn bf16 [p][256] (channel h*32+d)
// ---------------------------------------------------------------------------
__global__ void sample_attn(const float* __restrict__ off, const float* __restrict__ awl,
                            const bf16_t* __restrict__ Vproj, bf16_t* __restrict__ attnb) {
    int blk = blockIdx.x;
    int win = blk >> 3, h = blk & 7;
    int b = win >> 6, wi = win & 63, wy = wi >> 3, wx = wi & 7;
    int s = threadIdx.x;          // 0..63

    __shared__ __bf16 Vl[3][64][32];          // 12 KB
    for (int i = 0; i < 96; ++i) {            // 6144 elements / 64 threads
        int e = i * 64 + s;
        int l = e >> 11, r = e & 2047, ss = r >> 5, d = r & 31;
        int p = wpos(b, wy, wx, ss);
        Vl[l][ss][d] = Vproj[(((size_t)l * 16384 + p) << 8) + (h << 5) + d];
    }
    __syncthreads();

    int p = wpos(b, wy, wx, s);
    // softmax over L*P = 12
    float logit[12];
    float mx = -1e30f;
#pragma unroll
    for (int i = 0; i < 12; ++i) {
        logit[i] = awl[(size_t)p * 96 + h * 12 + i];
        mx = fmaxf(mx, logit[i]);
    }
    float se = 0.f;
#pragma unroll
    for (int i = 0; i < 12; ++i) { logit[i] = __expf(logit[i] - mx); se += logit[i]; }
    float inv = 1.f / se;

    float acc[32];
#pragma unroll
    for (int d = 0; d < 32; ++d) acc[d] = 0.f;

    float refx = (float)(s & 7) * (1.f / 7.f);
    float refy = (float)(s >> 3) * (1.f / 7.f);

    for (int l = 0; l < 3; ++l) {
        for (int pp = 0; pp < 4; ++pp) {
            int ch = (h * 3 + l) * 4 + pp;
            float ox = off[(size_t)p * 192 + ch * 2 + 0];
            float oy = off[(size_t)p * 192 + ch * 2 + 1];
            float lx = fminf(fmaxf(refx + ox * 0.125f, 0.f), 1.f);
            float ly = fminf(fmaxf(refy + oy * 0.125f, 0.f), 1.f);
            // grid_sample align_corners=False + border clamp: px = clip(loc*8-0.5, 0, 7)
            float px = fminf(fmaxf(lx * 8.f - 0.5f, 0.f), 7.f);
            float py = fminf(fmaxf(ly * 8.f - 0.5f, 0.f), 7.f);
            float x0f = floorf(px), y0f = floorf(py);
            float wxv = px - x0f, wyv = py - y0f;
            int x0 = (int)x0f, y0 = (int)y0f;
            int x1 = x0 + 1 > 7 ? 7 : x0 + 1;
            int y1 = y0 + 1 > 7 ? 7 : y0 + 1;
            float a = logit[l * 4 + pp] * inv;
            float w00 = a * (1.f - wxv) * (1.f - wyv);
            float w01 = a * wxv * (1.f - wyv);
            float w10 = a * (1.f - wxv) * wyv;
            float w11 = a * wxv * wyv;
            const __bf16* r00 = Vl[l][y0 * 8 + x0];
            const __bf16* r01 = Vl[l][y0 * 8 + x1];
            const __bf16* r10 = Vl[l][y1 * 8 + x0];
            const __bf16* r11 = Vl[l][y1 * 8 + x1];
#pragma unroll
            for (int d = 0; d < 32; ++d)
                acc[d] += w00 * (float)r00[d] + w01 * (float)r01[d]
                        + w10 * (float)r10[d] + w11 * (float)r11[d];
        }
    }
#pragma unroll
    for (int d = 0; d < 32; ++d)
        attnb[((size_t)p << 8) + (h << 5) + d] = (bf16_t)acc[d];
}

// ---------------------------------------------------------------------------
// Host launcher
// ---------------------------------------------------------------------------
extern "C" void kernel_launch(void* const* d_in, const int* in_sizes, int n_in,
                              void* d_out, int out_size, void* d_ws, size_t ws_size,
                              hipStream_t stream) {
    (void)in_sizes; (void)n_in; (void)out_size;
    const float* query  = (const float*)d_in[0];
    /* keys d_in[1] unused by reference */
    const float* values = (const float*)d_in[2];
    const float* so_w1  = (const float*)d_in[3];
    const float* so_b1  = (const float*)d_in[4];
    const float* so_g   = (const float*)d_in[5];
    const float* so_be  = (const float*)d_in[6];
    const float* so_w2  = (const float*)d_in[7];
    const float* so_b2  = (const float*)d_in[8];
    const float* aw_w1  = (const float*)d_in[9];
    const float* aw_b1  = (const float*)d_in[10];
    const float* aw_g   = (const float*)d_in[11];
    const float* aw_be  = (const float*)d_in[12];
    const float* aw_w2  = (const float*)d_in[13];
    const float* aw_b2  = (const float*)d_in[14];
    const float* vp_w   = (const float*)d_in[15];
    const float* vp_b   = (const float*)d_in[16];
    const float* op_w   = (const float*)d_in[17];
    const float* op_b   = (const float*)d_in[18];
    const float* lemb   = (const float*)d_in[19];

    // ----- workspace layout (bytes) -----
    // pk (packed bf16 weights)          : 0        .. 512K
    // Hso fp32 [16384][128]             : 512K     (+8M)   -> reused as off [16384][192] fp32
    // Haw fp32 [16384][128]             : 512K+8M  (+8M)
    // Hbso bf16 [16384][128]            : 512K+16M (+4M)   -> reused as attn bf16 [16384][256]
    // Hbaw bf16 [16384][128]            : 512K+20M (+4M)
    // awout fp32 [16384][96]            : 512K+24M (+6M)
    // Vproj bf16 [3][16384][256]        : 512K+30M (+24M)
    const size_t NEED = 524288ull + 16777216ull + 8388608ull + 6291456ull + 25165824ull;
    if (ws_size < NEED) return;

    char* ws = (char*)d_ws;
    bf16_t* pk     = (bf16_t*)ws;
    bf16_t* pk_so1 = pk;                 // 256x128 -> 32768
    bf16_t* pk_aw1 = pk + 32768;         // 256x128
    bf16_t* pk_so2 = pk + 65536;         // 128x192 -> 24576
    bf16_t* pk_aw2 = pk + 90112;         // 128x96  -> 12288
    bf16_t* pk_vp  = pk + 102400;        // 256x256 -> 65536
    bf16_t* pk_op  = pk + 167936;        // 256x256

    float*  Hso   = (float*)(ws + 524288);
    float*  Haw   = (float*)(ws + 524288 + 8388608);
    float*  offb  = Hso;                                   // reuse after GN
    bf16_t* Hbso  = (bf16_t*)(ws + 524288 + 16777216);
    bf16_t* Hbaw  = (bf16_t*)(ws + 524288 + 16777216 + 4194304);
    bf16_t* attnb = Hbso;                                  // reuse after stage-2
    float*  awout = (float*)(ws + 524288 + 16777216 + 8388608);
    bf16_t* Vproj = (bf16_t*)(ws + 524288 + 16777216 + 8388608 + 6291456);

    // 1) pack weights into WMMA B-fragment order
    pack_weight<<<64,  32, 0, stream>>>(so_w1, pk_so1, 256, 128);
    pack_weight<<<64,  32, 0, stream>>>(aw_w1, pk_aw1, 256, 128);
    pack_weight<<<48,  32, 0, stream>>>(so_w2, pk_so2, 128, 192);
    pack_weight<<<24,  32, 0, stream>>>(aw_w2, pk_aw2, 128, 96);
    pack_weight<<<128, 32, 0, stream>>>(vp_w,  pk_vp,  256, 256);
    pack_weight<<<128, 32, 0, stream>>>(op_w,  pk_op,  256, 256);

    // 2) stage-1 GEMMs directly from [C][H][W] slices of query
    //    N=128, NB=4 -> 256 mTiles * 2 nGroups = 512 waves -> 128 blocks
    for (int b = 0; b < 4; ++b) {
        const float* q = query + (size_t)b * 256 * 4096;
        gemm_chw<4,false><<<128, 128, 0, stream>>>(q, nullptr, pk_so1, so_b1, Hso, nullptr, b * 4096, 128);
        gemm_chw<4,false><<<128, 128, 0, stream>>>(q, nullptr, pk_aw1, aw_b1, Haw, nullptr, b * 4096, 128);
    }

    // 3) per-window GroupNorm + ReLU -> bf16
    gn_relu<<<256, 128, 0, stream>>>(Hso, Haw, so_g, so_be, aw_g, aw_be, Hbso, Hbaw);

    // 4) stage-2 GEMMs (offsets + attention logits); off overlays dead H region
    //    so2: N=192, NB=4 -> 1024*3 = 3072 waves -> 768 blocks
    //    aw2: N=96,  NB=2 -> 1024*3 = 3072 waves -> 768 blocks
    gemm_rowA<4,false><<<768, 128, 0, stream>>>(Hbso, pk_so2, so_b2, offb,  nullptr, 16384, 128, 192);
    gemm_rowA<2,false><<<768, 128, 0, stream>>>(Hbaw, pk_aw2, aw_b2, awout, nullptr, 16384, 128, 96);

    // 5) value projection per (b, level), level_embed folded into A-load; bf16 out
    //    N=256, NB=4 -> 256*4 = 1024 waves -> 256 blocks per slice
    for (int b = 0; b < 4; ++b)
        for (int l = 0; l < 3; ++l) {
            const float* v = values + (size_t)(b * 3 + l) * 256 * 4096;
            gemm_chw<4,true><<<256, 128, 0, stream>>>(v, lemb + l * 256, pk_vp, vp_b,
                                                      nullptr, Vproj, l * 16384 + b * 4096, 256);
        }

    // 6) softmax + bilinear sampling + attention combine (attnb overlays dead Hb)
    sample_attn<<<2048, 64, 0, stream>>>(offb, awout, Vproj, attnb);

    // 7) output projection -> d_out [4][256][64][64] fp32
    //    N=256, NB=4 -> 1024*4 = 4096 waves -> 1024 blocks
    gemm_op<<<1024, 128, 0, stream>>>(attnb, pk_op, op_b, (float*)d_out);
}
